// DotScaledAttention_171798692107
// MI455X (gfx1250) — matrix operand order repaired and checked by reading, not compiled, not run on hardware
//
#include <hip/hip_runtime.h>
#include <hip/hip_bf16.h>
#include <cmath>

// D = A x B + C WMMA types (CDNA5 gfx1250, wave32)
typedef __attribute__((ext_vector_type(16))) __bf16 v16bf;
typedef __attribute__((ext_vector_type(8)))  float  v8f;

__device__ __forceinline__ unsigned short f2bf(float f) {
    union { float f; unsigned u; } v; v.f = f;
    unsigned r = v.u + 0x7FFFu + ((v.u >> 16) & 1u);  // round-to-nearest-even
    return (unsigned short)(r >> 16);
}

// ---------------------------------------------------------------------------
// Projection GEMM: out[M,Nn] (bf16) = (X[M,K] @ W[K,Nn] + bias) * scale
// transposed==1 stores out as [B, Nn, seqN] (d-major) for the V matrix.
// Block: 128 threads (4 waves). Tile: 64(M) x 64(N), k-chunk 32.
// ---------------------------------------------------------------------------
#define MT 64
#define NT 64
#define KT 32

__global__ __launch_bounds__(128) void proj_gemm_kernel(
    const float* __restrict__ X, const float* __restrict__ W,
    const float* __restrict__ bias, unsigned short* __restrict__ out,
    int M, int Kd, int Nn, float scale, int transposed, int seqN)
{
    __shared__ unsigned short sA [MT][KT];   // A tile, bf16, row-major
    __shared__ unsigned short sBT[NT][KT];   // W tile, bf16, transposed (N x K)

    const int tid  = threadIdx.x;
    const int wave = tid >> 5;
    const int lane = tid & 31;
    const int lcol = lane & 15;
    const int lhalf = lane >> 4;

    const int m0 = blockIdx.x * MT;
    const int n0 = blockIdx.y * NT;

    v8f c[4] = {};

    const int kbA = lhalf ? 16 : 0;    // A-fragment K base (contiguous k = 16h + s)
    const int kbB = lhalf ? 16 : 0;    // B-fragment K base

    for (int k0 = 0; k0 < Kd; k0 += KT) {
        // stage A: 64x32 fp32 -> bf16, vectorized float4 loads, 64-bit (uint2) LDS stores
        #pragma unroll
        for (int i = tid; i < MT * KT / 4; i += 128) {
            const int r  = i >> 3;          // 8 float4 groups per row
            const int c4 = (i & 7) * 4;
            const float4 f = *(const float4*)&X[(size_t)(m0 + r) * Kd + k0 + c4];
            const unsigned lo = (unsigned)f2bf(f.x) | ((unsigned)f2bf(f.y) << 16);
            const unsigned hi = (unsigned)f2bf(f.z) | ((unsigned)f2bf(f.w) << 16);
            *(uint2*)&sA[r][c4] = make_uint2(lo, hi);
        }
        // stage W transposed: float4 loads along N, scattered bf16 stores
        #pragma unroll
        for (int i = tid; i < NT * KT / 4; i += 128) {
            const int r  = i >> 4;          // 16 groups of 4 n per K-row
            const int c4 = (i & 15) * 4;
            const float4 f = *(const float4*)&W[(size_t)(k0 + r) * Nn + n0 + c4];
            sBT[c4 + 0][r] = f2bf(f.x);
            sBT[c4 + 1][r] = f2bf(f.y);
            sBT[c4 + 2][r] = f2bf(f.z);
            sBT[c4 + 3][r] = f2bf(f.w);
        }
        __syncthreads();

        // A fragment: 16 rows of this wave, K chunk (two 16B LDS loads)
        v16bf a;
        {
            const int arow = wave * 16 + lcol;
            ((uint4*)&a)[0] = *(const uint4*)&sA[arow][kbA];
            ((uint4*)&a)[1] = *(const uint4*)&sA[arow][kbA + 8];
        }
        #pragma unroll
        for (int t = 0; t < 4; ++t) {
            v16bf b;
            const int col = t * 16 + lcol;
            ((uint4*)&b)[0] = *(const uint4*)&sBT[col][kbB];
            ((uint4*)&b)[1] = *(const uint4*)&sBT[col][kbB + 8];
            c[t] = __builtin_amdgcn_wmma_f32_16x16x32_bf16(
                       false, a, false, b, (short)0, c[t], false, false);
        }
        __syncthreads();
    }

    // epilogue: bias + scale, bf16 store
    #pragma unroll
    for (int t = 0; t < 4; ++t) {
        const int n = n0 + t * 16 + lcol;
        const float bv = bias[n];
        #pragma unroll
        for (int r = 0; r < 8; ++r) {
            const int m = m0 + wave * 16 + 8 * lhalf + r;   // C layout row
            const unsigned short h = f2bf((c[t][r] + bv) * scale);
            if (!transposed) {
                out[(size_t)m * Nn + n] = h;
            } else {
                const int b = m / seqN, s = m - b * seqN;
                out[((size_t)b * Nn + n) * seqN + s] = h;    // [B, d, N]
            }
        }
    }
}

// ---------------------------------------------------------------------------
// Flash attention (causal): one 16-query tile per 128-thread block.
// 4 waves split d=512 into 128-wide slices for PV/output; score tile 16x32
// is computed redundantly per wave (K fragments come straight from L2/WGP$).
// ---------------------------------------------------------------------------
__global__ __launch_bounds__(128) void flash_attn_kernel(
    const unsigned short* __restrict__ Q,   // [B*N, 512] bf16 row-major (pre-scaled)
    const unsigned short* __restrict__ K,   // [B*N, 512] bf16 row-major
    const unsigned short* __restrict__ VT,  // [B, 512, N] bf16 d-major
    float* __restrict__ out,                // [B, N, 512] fp32
    int N, int D)
{
    __shared__ unsigned short sQ[16][512];      // 16 KB
    __shared__ unsigned short sP[4][16][32];    // per-wave P tile, 4 KB

    const int tid  = threadIdx.x;
    const int wave = tid >> 5;
    const int lane = tid & 31;
    const int lcol = lane & 15;
    const int lhalf = lane >> 4;

    const int b  = blockIdx.y;
    const int q0 = blockIdx.x * 16;

    const unsigned short* Qb = Q + ((size_t)b * N + q0) * D;
    const unsigned short* Kb = K + (size_t)b * N * D;
    const unsigned short* Vb = VT + (size_t)b * D * N;

    // ---- stage Q tile (contiguous 16 KB) ----
#if __has_builtin(__builtin_amdgcn_global_load_async_to_lds_b128)
    {   // async DMA into LDS (ASYNCcnt-tracked), 128 lanes x 16B x 8 iters
        typedef int v4i __attribute__((vector_size(16)));
        typedef __attribute__((address_space(1))) v4i* gp_t;
        typedef __attribute__((address_space(3))) v4i* lp_t;
        #pragma unroll
        for (int it = 0; it < 8; ++it) {
            const int off = (it * 128 + tid) * 8;   // ushorts (16B per thread)
            __builtin_amdgcn_global_load_async_to_lds_b128(
                (gp_t)(Qb + off), (lp_t)(&sQ[0][0] + off), 0, 0);
        }
        asm volatile("s_wait_asynccnt 0" ::: "memory");
    }
#else
    for (int i = tid * 4; i < 16 * 512; i += 128 * 4)
        *(uint2*)&sQ[0][i] = *(const uint2*)&Qb[i];
#endif
    __syncthreads();

    v8f  acc[8] = {};
    float mrun[8], lrun[8];
    #pragma unroll
    for (int r = 0; r < 8; ++r) { mrun[r] = -INFINITY; lrun[r] = 0.0f; }

    const int dw0 = wave * 128;                 // this wave's d slice
    const int kbA = lhalf ? 16 : 0;             // A K base (contiguous k = 16h + s)
    const int dB  = lhalf ? 16 : 0;             // B-layout K-row base
    const int nchunks = (q0 + 16 + 31) / 32;    // causal: keys <= q0+15

    for (int ch = 0; ch < nchunks; ++ch) {
        const int k0 = ch * 32;

        // prefetch next chunk's K rows / V slice (global_prefetch_b8)
        if (ch + 1 < nchunks) {
            __builtin_prefetch(Kb + (size_t)(k0 + 32 + lane) * D, 0, 0);
            __builtin_prefetch(Vb + (size_t)(dw0 + lane * 4) * N + k0 + 32, 0, 0);
        }

        // ---- S = Q . K^T over d=512, two 16x16 key sub-tiles ----
        v8f s0 = {}, s1 = {};
        for (int d0 = 0; d0 < 512; d0 += 32) {
            v16bf a;
            ((uint4*)&a)[0] = *(const uint4*)&sQ[lcol][d0 + kbA];
            ((uint4*)&a)[1] = *(const uint4*)&sQ[lcol][d0 + kbA + 8];
            const unsigned short* kp0 = Kb + (size_t)(k0 + lcol) * D + d0 + dB;
            const unsigned short* kp1 = Kb + (size_t)(k0 + 16 + lcol) * D + d0 + dB;
            v16bf b0, b1;
            ((uint4*)&b0)[0] = *(const uint4*)kp0;
            ((uint4*)&b0)[1] = *(const uint4*)(kp0 + 8);
            ((uint4*)&b1)[0] = *(const uint4*)kp1;
            ((uint4*)&b1)[1] = *(const uint4*)(kp1 + 8);
            s0 = __builtin_amdgcn_wmma_f32_16x16x32_bf16(
                     false, a, false, b0, (short)0, s0, false, false);
            s1 = __builtin_amdgcn_wmma_f32_16x16x32_bf16(
                     false, a, false, b1, (short)0, s1, false, false);
        }

        // ---- causal mask + online softmax (rows live in fixed lane halves) ----
        #pragma unroll
        for (int r = 0; r < 8; ++r) {
            const int row = q0 + 8 * lhalf + r;
            const int c0 = k0 + lcol, c1 = k0 + 16 + lcol;
            float v0 = (c0 <= row) ? s0[r] : -INFINITY;
            float v1 = (c1 <= row) ? s1[r] : -INFINITY;
            float mx = fmaxf(v0, v1);
            mx = fmaxf(mx, __shfl_xor(mx, 1));
            mx = fmaxf(mx, __shfl_xor(mx, 2));
            mx = fmaxf(mx, __shfl_xor(mx, 4));
            mx = fmaxf(mx, __shfl_xor(mx, 8));
            const float mnew = fmaxf(mrun[r], mx);
            const float corr = __expf(mrun[r] - mnew);
            const float p0 = __expf(v0 - mnew);
            const float p1 = __expf(v1 - mnew);
            float rs = p0 + p1;
            rs += __shfl_xor(rs, 1);
            rs += __shfl_xor(rs, 2);
            rs += __shfl_xor(rs, 4);
            rs += __shfl_xor(rs, 8);
            lrun[r] = lrun[r] * corr + rs;
            mrun[r] = mnew;
            s0[r] = p0; s1[r] = p1;
            #pragma unroll
            for (int t = 0; t < 8; ++t) acc[t][r] *= corr;
        }

        // ---- re-layout P (C layout -> A layout) through per-wave LDS ----
        #pragma unroll
        for (int r = 0; r < 8; ++r) {
            sP[wave][8 * lhalf + r][lcol]      = f2bf(s0[r]);
            sP[wave][8 * lhalf + r][16 + lcol] = f2bf(s1[r]);
        }
        v16bf pa;   // wave-private region; DScnt dependency handled by compiler
        ((uint4*)&pa)[0] = *(const uint4*)&sP[wave][lcol][kbA];
        ((uint4*)&pa)[1] = *(const uint4*)&sP[wave][lcol][kbA + 8];

        // ---- acc += P (16x32) x V (32 x 128-slice), VT is d-major ----
        #pragma unroll
        for (int t = 0; t < 8; ++t) {
            const int dcol = dw0 + t * 16 + lcol;
            const unsigned short* vp = Vb + (size_t)dcol * N + k0 + dB;
            v16bf vb;
            ((uint4*)&vb)[0] = *(const uint4*)vp;
            ((uint4*)&vb)[1] = *(const uint4*)(vp + 8);
            acc[t] = __builtin_amdgcn_wmma_f32_16x16x32_bf16(
                         false, pa, false, vb, (short)0, acc[t], false, false);
        }
    }

    // ---- normalize and store fp32 output ----
    #pragma unroll
    for (int t = 0; t < 8; ++t) {
        const int dcol = dw0 + t * 16 + lcol;
        #pragma unroll
        for (int r = 0; r < 8; ++r) {
            const int row = q0 + 8 * lhalf + r;
            out[((size_t)b * N + row) * D + dcol] = acc[t][r] / lrun[r];
        }
    }
}

// ---------------------------------------------------------------------------
// Host launcher
// ---------------------------------------------------------------------------
extern "C" void kernel_launch(void* const* d_in, const int* in_sizes, int n_in,
                              void* d_out, int out_size, void* d_ws, size_t ws_size,
                              hipStream_t stream) {
    (void)in_sizes; (void)n_in; (void)out_size; (void)ws_size;
    const float* x0 = (const float*)d_in[0];
    const float* x1 = (const float*)d_in[1];
    const float* x2 = (const float*)d_in[2];
    const float* Wq = (const float*)d_in[3];
    const float* bq = (const float*)d_in[4];
    const float* Wk = (const float*)d_in[5];
    const float* bk = (const float*)d_in[6];
    float* out = (float*)d_out;

    const int B = 8, N = 2048, DE = 1024, DR = 512;
    const int M = B * N;  // 16384
    const float scale = 1.0f / sqrtf((float)DR);

    // workspace: Q, K (row-major bf16) + V^T (d-major bf16) = 48 MB
    unsigned short* Qw = (unsigned short*)d_ws;
    unsigned short* Kw = Qw + (size_t)M * DR;
    unsigned short* Vw = Kw + (size_t)M * DR;

    dim3 gg(M / MT, DR / NT);   // (256, 8)
    proj_gemm_kernel<<<gg, 128, 0, stream>>>(x0, Wq, bq, Qw, M, DE, DR, scale, 0, N);
    proj_gemm_kernel<<<gg, 128, 0, stream>>>(x1, Wk, bk, Kw, M, DE, DR, 1.0f, 0, N);
    proj_gemm_kernel<<<gg, 128, 0, stream>>>(x2, Wk, bk, Vw, M, DE, DR, 1.0f, 1, N);

    dim3 ga(N / 16, B);         // (128, 8)
    flash_attn_kernel<<<ga, 128, 0, stream>>>(Qw, Kw, Vw, out, N, DR);
}
